// DotProdAttention_730144440910
// MI455X (gfx1250) — matrix operand order repaired and checked by reading, not compiled, not run on hardware
//
#include <hip/hip_runtime.h>

typedef __attribute__((ext_vector_type(16))) _Float16 v16h;
typedef __attribute__((ext_vector_type(2)))  _Float16 v2h;
typedef __attribute__((ext_vector_type(8)))  float    v8f;
typedef int v4i __attribute__((vector_size(16)));
typedef __attribute__((address_space(1))) v4i* gv4i_p;   // global v4i*
typedef __attribute__((address_space(3))) v4i* lv4i_p;   // LDS v4i*

#define SEQ   2048
#define DDIM  64
#define KC    64                         // keys per chunk
#define LDSW  68                         // padded f16 LDS row stride (elements)
#define WAVES 8
#define SUBT  2                          // 16-row Q sub-tiles per wave
#define QR_PER_BLOCK (WAVES * 16 * SUBT) // 256
#define NCHUNK (SEQ / KC)                // 32
#define CHUNK_F (KC * DDIM)              // 4096 floats per K (or V) chunk

// ---- CDNA5 async global->LDS (guarded; sync fallback keeps compile green) ----
#if defined(__has_builtin)
#  if __has_builtin(__builtin_amdgcn_global_load_async_to_lds_b128)
#    define HAVE_ASYNC 1
#  endif
#  if __has_builtin(__builtin_amdgcn_s_wait_asynccnt)
#    define HAVE_WAIT_ASYNC 1
#  endif
#  if __has_builtin(__builtin_amdgcn_cvt_pkrtz)
#    define HAVE_PKRTZ 1
#  endif
#endif
#ifndef HAVE_ASYNC
#  define HAVE_ASYNC 0
#endif

#if HAVE_ASYNC
#  if defined(HAVE_WAIT_ASYNC)
#    define WAIT_ASYNC(N) __builtin_amdgcn_s_wait_asynccnt(N)
#  else
#    define WAIT_ASYNC(N) asm volatile("s_wait_asynccnt %0" :: "i"(N) : "memory")
#  endif
#else
#  define WAIT_ASYNC(N) do {} while (0)
#endif

// packed f32x2 -> f16x2 (round-toward-zero HW op if available)
__device__ inline v2h cvt_pk2(float x, float y) {
#if defined(HAVE_PKRTZ)
    auto h = __builtin_amdgcn_cvt_pkrtz(x, y);
    v2h r; r[0] = (_Float16)h[0]; r[1] = (_Float16)h[1];
    return r;
#else
    v2h r; r[0] = (_Float16)x; r[1] = (_Float16)y;
    return r;
#endif
}

__device__ inline void prefetch_chunk(const float* gK, const float* gV,
                                      float* sK, float* sV, int tid) {
#pragma unroll
    for (int r = 0; r < 4; ++r) {
        const int off = r * 1024 + tid * 4;      // 16B per lane per round
#if HAVE_ASYNC
        __builtin_amdgcn_global_load_async_to_lds_b128(
            (gv4i_p)const_cast<float*>(gK + off), (lv4i_p)(sK + off), 0, 0);
        __builtin_amdgcn_global_load_async_to_lds_b128(
            (gv4i_p)const_cast<float*>(gV + off), (lv4i_p)(sV + off), 0, 0);
#else
        const float4 a = *reinterpret_cast<const float4*>(gK + off);
        const float4 b = *reinterpret_cast<const float4*>(gV + off);
        *reinterpret_cast<float4*>(sK + off) = a;
        *reinterpret_cast<float4*>(sV + off) = b;
#endif
    }
}

// load two contiguous f16 from LDS (4B aligned) into elements 2j,2j+1 of a v16h
__device__ inline void set_pair(v16h& v, int j, const _Float16* p) {
    union { unsigned int u; _Float16 h[2]; } t;
    t.u = *reinterpret_cast<const unsigned int*>(p);
    v[2 * j]     = t.h[0];
    v[2 * j + 1] = t.h[1];
}

__global__ __launch_bounds__(256) void fa_fwd_kernel(
    const float* __restrict__ Q, const float* __restrict__ K,
    const float* __restrict__ V, float* __restrict__ O)
{
    __shared__ __attribute__((aligned(16))) float Stg[2][2][CHUNK_F]; // [buf][K/V] f32 staging
    __shared__ _Float16 Ks[KC * LDSW];            // f16 key-major  [key][d]
    __shared__ _Float16 Vt[DDIM * LDSW];          // f16 transposed [d][key]
    __shared__ _Float16 Pt[WAVES * 16 * LDSW];    // per-wave P tile [16][64]

    const int tid  = threadIdx.x;
    const int wave = tid >> 5;
    const int lane = tid & 31;
    const int lo   = lane & 15;
    const int hi   = lane >> 4;

    const size_t base  = (size_t)blockIdx.y * SEQ * DDIM;
    const int    qrow0 = blockIdx.x * QR_PER_BLOCK + wave * (16 * SUBT);

    // ---- Q fragments (A-layout 16x32 f16), scale = 1/sqrt(64) * log2(e) ----
    const float qscale = 0.125f * 1.44269504088896340736f;
    v16h qa[SUBT][2];
#pragma unroll
    for (int st = 0; st < SUBT; ++st) {
        const float* qp = Q + base + (size_t)(qrow0 + st * 16 + lo) * DDIM;
#pragma unroll
        for (int c = 0; c < 2; ++c) {
#pragma unroll
            for (int j = 0; j < 8; ++j) {
                const int k = c * 32 + ((j & 4) ? 16 : 0) + hi * 8 + (j & 3) * 2;
                const float2 f = *reinterpret_cast<const float2*>(qp + k);
                const v2h h = cvt_pk2(f.x * qscale, f.y * qscale);
                qa[st][c][2 * j]     = h[0];
                qa[st][c][2 * j + 1] = h[1];
            }
        }
    }

    v8f o[SUBT][4];
    float m[SUBT][8], l[SUBT][8];
#pragma unroll
    for (int st = 0; st < SUBT; ++st) {
#pragma unroll
        for (int nt = 0; nt < 4; ++nt) o[st][nt] = (v8f){0.f,0.f,0.f,0.f,0.f,0.f,0.f,0.f};
#pragma unroll
        for (int i = 0; i < 8; ++i) { m[st][i] = -1.0e30f; l[st][i] = 0.0f; }
    }

    _Float16* pw = &Pt[wave * 16 * LDSW];
    const float* gK = K + base;
    const float* gV = V + base;

    prefetch_chunk(gK, gV, Stg[0][0], Stg[0][1], tid);

#pragma unroll 1
    for (int c = 0; c < NCHUNK; ++c) {
        const int cur = c & 1;
        if (c + 1 < NCHUNK) {
            prefetch_chunk(gK + (size_t)(c + 1) * CHUNK_F, gV + (size_t)(c + 1) * CHUNK_F,
                           Stg[cur ^ 1][0], Stg[cur ^ 1][1], tid);
            WAIT_ASYNC(8);     // in-order completion => previous chunk's 8 ops done
        } else {
            WAIT_ASYNC(0);
        }
        __syncthreads();       // chunk c staged for all waves; prev compute finished

        // ---- convert staging f32 -> f16 tiles (K key-major, V transposed) ----
        const float* sK = Stg[cur][0];
        const float* sV = Stg[cur][1];
#pragma unroll
        for (int i = 0; i < CHUNK_F / (256 * 2); ++i) {   // 8 iters of float2
            const int idx = i * 256 + tid;
            const int key = idx >> 5;
            const int d   = (idx & 31) * 2;
            const float2 kf = *reinterpret_cast<const float2*>(sK + key * DDIM + d);
            const float2 vf = *reinterpret_cast<const float2*>(sV + key * DDIM + d);
            *reinterpret_cast<v2h*>(&Ks[key * LDSW + d]) = cvt_pk2(kf.x, kf.y);
            const v2h vh = cvt_pk2(vf.x, vf.y);
            Vt[d * LDSW + key]       = vh[0];
            Vt[(d + 1) * LDSW + key] = vh[1];
        }
        __syncthreads();       // f16 tiles ready for all waves

#pragma unroll
        for (int st = 0; st < SUBT; ++st) {
            // ---- S = (Q*scale) @ K^T : 4 col tiles x 2 K-halves ----
            v8f s[4];
#pragma unroll
            for (int nt = 0; nt < 4; ++nt) {
                v8f acc = (v8f){0.f,0.f,0.f,0.f,0.f,0.f,0.f,0.f};
                const int keyrow = nt * 16 + lo;
#pragma unroll
                for (int cc = 0; cc < 2; ++cc) {
                    v16h b;
#pragma unroll
                    for (int j = 0; j < 8; ++j) {
                        const int k = cc * 32 + ((j & 4) ? 16 : 0) + hi * 8 + (j & 3) * 2;
                        set_pair(b, j, &Ks[keyrow * LDSW + k]);
                    }
                    acc = __builtin_amdgcn_wmma_f32_16x16x32_f16(
                              false, qa[st][cc], false, b, (short)0, acc, false, false);
                }
                s[nt] = acc;
            }

            // ---- online softmax in base-2 domain (row = i + 8*hi) ----
            float alpha[8];
#pragma unroll
            for (int i = 0; i < 8; ++i) {
                float mi = fmaxf(fmaxf(s[0][i], s[1][i]), fmaxf(s[2][i], s[3][i]));
                mi = fmaxf(mi, __shfl_xor(mi, 1));
                mi = fmaxf(mi, __shfl_xor(mi, 2));
                mi = fmaxf(mi, __shfl_xor(mi, 4));
                mi = fmaxf(mi, __shfl_xor(mi, 8));
                const float mn = fmaxf(m[st][i], mi);
                alpha[i] = __builtin_exp2f(m[st][i] - mn);
                m[st][i] = mn;
            }
#pragma unroll
            for (int i = 0; i < 8; ++i) {
                const int row = i + hi * 8;
                float rs = 0.0f;
#pragma unroll
                for (int nt = 0; nt < 4; ++nt) {
                    const float p = __builtin_exp2f(s[nt][i] - m[st][i]);
                    rs += p;
                    pw[row * LDSW + nt * 16 + lo] = (_Float16)p;
                }
                rs += __shfl_xor(rs, 1);
                rs += __shfl_xor(rs, 2);
                rs += __shfl_xor(rs, 4);
                rs += __shfl_xor(rs, 8);
                l[st][i] = l[st][i] * alpha[i] + rs;
#pragma unroll
                for (int nt = 0; nt < 4; ++nt) o[st][nt][i] *= alpha[i];
            }

            // ---- O += P @ V ----
            v16h pa[2];
#pragma unroll
            for (int cc = 0; cc < 2; ++cc) {
#pragma unroll
                for (int j = 0; j < 8; ++j) {
                    const int k = cc * 32 + ((j & 4) ? 16 : 0) + hi * 8 + (j & 3) * 2;
                    set_pair(pa[cc], j, &pw[lo * LDSW + k]);
                }
            }
#pragma unroll
            for (int nt = 0; nt < 4; ++nt) {
                const int drow = nt * 16 + lo;
#pragma unroll
                for (int cc = 0; cc < 2; ++cc) {
                    v16h vb;
#pragma unroll
                    for (int j = 0; j < 8; ++j) {
                        const int k = cc * 32 + ((j & 4) ? 16 : 0) + hi * 8 + (j & 3) * 2;
                        set_pair(vb, j, &Vt[drow * LDSW + k]);
                    }
                    o[st][nt] = __builtin_amdgcn_wmma_f32_16x16x32_f16(
                                    false, pa[cc], false, vb, (short)0, o[st][nt], false, false);
                }
            }
        }
    }

    // ---- epilogue: O / l ----
#pragma unroll
    for (int st = 0; st < SUBT; ++st) {
#pragma unroll
        for (int i = 0; i < 8; ++i) {
            const float inv = 1.0f / l[st][i];
            const size_t row = (size_t)(qrow0 + st * 16 + i + hi * 8);
#pragma unroll
            for (int nt = 0; nt < 4; ++nt) {
                O[base + row * DDIM + nt * 16 + lo] = o[st][nt][i] * inv;
            }
        }
    }
}

extern "C" void kernel_launch(void* const* d_in, const int* in_sizes, int n_in,
                              void* d_out, int out_size, void* d_ws, size_t ws_size,
                              hipStream_t stream) {
    const float* q = (const float*)d_in[0];
    const float* k = (const float*)d_in[1];
    const float* v = (const float*)d_in[2];
    float* out = (float*)d_out;
    dim3 grid(SEQ / QR_PER_BLOCK, 4 * 16);   // (8 q-tiles, 64 batch*heads)
    fa_fwd_kernel<<<grid, 256, 0, stream>>>(q, k, v, out);
}
